// SCUNet_16484084483158
// MI455X (gfx1250) — matrix operand order repaired and checked by reading, not compiled, not run on hardware
//
#include <hip/hip_runtime.h>
#include <hip/hip_bf16.h>

// ---------------------------------------------------------------------------
// Fused shifted-window attention (SCUNet block) for gfx1250 (MI455X).
//   grid = 4096 workgroups (batch 4 x 1024 windows), 256 threads = 8 waves.
//   Fully fused through LDS: only x (268 MB) read / out (268 MB) written to
//   HBM => ~23 us at 23.3 TB/s. Weights (0.5 MB f16) are L2-resident.
//   Matrix math: v_wmma_f32_16x16x32_f16 (K=32 == head_dim).
//   V is stored transposed in LDS so every WMMA operand load is K-contiguous
//   (merges into ds_load_b128 / global_load_b128).
// ---------------------------------------------------------------------------

typedef __attribute__((ext_vector_type(16))) _Float16 v16h;
typedef __attribute__((ext_vector_type(8)))  float    v8f;
typedef __attribute__((ext_vector_type(2)))  _Float16 v2h;

#define IMG      256
#define CH       256
#define P2       64          // tokens per window
#define NHEAD    8
#define HDIM     32
#define QK_LD    520         // 512 + pad (q|k per token)
#define VT_LD    72          // 64 + pad  (v transposed: [d][token])
#define X_LD     264         // 256 + pad (x window / attn-out)
#define P_LD     72          // 64 + pad  (softmax P tile)
#define NEG_INF  (-1.0e30f)

// ---- fragment loaders (gfx1250 wave32 WMMA 16x16x32 f16 layouts) ----------
// A (16x32): lane L holds row M=L%16; halves 0..7 -> K=e(+8 if L>=16),
//            halves 8..15 -> K=16+e'(+8 if L>=16). K pairs contiguous.
__device__ __forceinline__ v16h load_a_frag(const _Float16* base, int row0,
                                            int stride, int kbase, int lane) {
  const _Float16* rp = base + (size_t)(row0 + (lane & 15)) * stride
                            + kbase + ((lane & 16) ? 8 : 0);
  v16h a;
#pragma unroll
  for (int i = 0; i < 4; ++i) {
    v2h p0 = *(const v2h*)(rp + 2 * i);        // K = 2i, 2i+1
    v2h p1 = *(const v2h*)(rp + 16 + 2 * i);   // K = 16+2i, 16+2i+1
    a[2 * i]     = p0[0]; a[2 * i + 1]     = p0[1];
    a[8 + 2 * i] = p1[0]; a[8 + 2 * i + 1] = p1[1];
  }
  return a;
}

// B (32x16) from a matrix stored [N][K] (K contiguous): element(K,N) =
// mat[(nbase+lane%16)*ld + kbase + K],  same K order as A: h<8 -> K=h(+8 if
// lane>=16), h>=8 -> K=16+(h-8)(+8 if lane>=16).
__device__ __forceinline__ v16h load_bT_frag(const _Float16* mat, int nbase,
                                             int ld, int kbase, int lane) {
  const _Float16* rp = mat + (size_t)(nbase + (lane & 15)) * ld
                           + kbase + ((lane & 16) ? 8 : 0);
  v16h b;
#pragma unroll
  for (int i = 0; i < 4; ++i) {
    v2h p0 = *(const v2h*)(rp + 2 * i);
    v2h p1 = *(const v2h*)(rp + 16 + 2 * i);
    b[2 * i] = p0[0]; b[2 * i + 1] = p0[1];
    b[8 + 2 * i] = p1[0]; b[8 + 2 * i + 1] = p1[1];
  }
  return b;
}

__device__ __forceinline__ v8f wmma_f16(v16h a, v16h b, v8f c) {
  return __builtin_amdgcn_wmma_f32_16x16x32_f16(false, a, false, b,
                                                (short)0, c, false, false);
}

// ---- weight fp32 -> f16 pre-conversion ------------------------------------
__global__ void cvt_f32_f16(const float* __restrict__ src,
                            _Float16* __restrict__ dst, int n) {
  int i = blockIdx.x * 256 + threadIdx.x;
  if (i < n) dst[i] = (_Float16)src[i];
}

// ---- fused window-attention kernel ----------------------------------------
__global__ __launch_bounds__(256)
void swin_fused(const float* __restrict__ x,
                const _Float16* __restrict__ wqkv_h,   // [768][256] f16
                const float* __restrict__ b_qkv,       // [768]
                const float* __restrict__ rel_pos,     // [8][15][15]
                const _Float16* __restrict__ wout_h,   // [256][256] f16
                const float* __restrict__ b_out,       // [256]
                float* __restrict__ out) {
  __shared__ _Float16 lds_qk[P2][QK_LD];         // q|k per token      66.5 KB
  __shared__ _Float16 lds_vT[CH][VT_LD];         // v transposed       36.9 KB
  __shared__ _Float16 lds_x[P2][X_LD];           // x window/attn-out  33.8 KB
  __shared__ _Float16 p_buf[NHEAD][16][P_LD];    // per-wave P tile    18.4 KB
  __shared__ float    lds_rel[NHEAD * 225];      // rel-pos bias        7.2 KB

  const int tid   = threadIdx.x;
  const int lane  = tid & 31;
  const int wave  = tid >> 5;                    // 0..7
  const int wg    = blockIdx.x;                  // 0..4095
  const int b_idx = wg >> 10;
  const int n_idx = wg & 1023;
  const int wy = n_idx >> 5, wx = n_idx & 31;
  const int half16 = (lane & 16) ? 8 : 0;
  const int q_n    = lane & 15;

  // ---- Stage A: load rolled window -> LDS (f16), fully coalesced ----------
  {
    const int t0 = wave * 8;
#pragma unroll
    for (int i = 0; i < 8; ++i) {
      const int t  = t0 + i;
      const int ty = t >> 3, tx = t & 7;
      const int row = (wy * 8 + ty + 4) & 255;   // roll(-4,-4)
      const int col = (wx * 8 + tx + 4) & 255;
      const float4* xp = (const float4*)(x +
          ((((size_t)b_idx * IMG + row) * IMG + col) * CH));
      float4 f0 = xp[lane];                      // 512B contiguous / wave
      float4 f1 = xp[32 + lane];
      const int c0 = lane * 4;
      lds_x[t][c0 + 0] = (_Float16)f0.x;  lds_x[t][c0 + 1] = (_Float16)f0.y;
      lds_x[t][c0 + 2] = (_Float16)f0.z;  lds_x[t][c0 + 3] = (_Float16)f0.w;
      lds_x[t][128 + c0 + 0] = (_Float16)f1.x;
      lds_x[t][128 + c0 + 1] = (_Float16)f1.y;
      lds_x[t][128 + c0 + 2] = (_Float16)f1.z;
      lds_x[t][128 + c0 + 3] = (_Float16)f1.w;
    }
    for (int i = tid; i < NHEAD * 225; i += 256) lds_rel[i] = rel_pos[i];
  }
  __syncthreads();

  // ---- Stage B: QKV projection [64,256] x [256,768]^T ---------------------
  // 2 waves per M-tile; A fragments hoisted (loaded once per wave).
  {
    const int Mt  = wave >> 1;
    const int rb  = Mt * 16 + half16;
    v16h aX[8];
#pragma unroll
    for (int kc = 0; kc < 8; ++kc)
      aX[kc] = load_a_frag(&lds_x[0][0], Mt * 16, X_LD, kc * 32, lane);

    const int nt0 = (wave & 1) * 24;
    for (int Nt = nt0; Nt < nt0 + 24; ++Nt) {    // 48 N-tiles of 16
      v8f acc = {};
#pragma unroll
      for (int kc = 0; kc < 8; ++kc) {
        v16h b = load_bT_frag(wqkv_h, Nt * 16, 256, kc * 32, lane);
        acc = wmma_f16(aX[kc], b, acc);
      }
      const int o  = Nt * 16 + q_n;
      const float bv = b_qkv[o];
      if (Nt < 32) {                             // q | k : row-major
#pragma unroll
        for (int r = 0; r < 8; ++r)
          lds_qk[rb + r][o] = (_Float16)(acc[r] + bv);
      } else {                                   // v : transposed [d][token]
        const int d = o - 512;
#pragma unroll
        for (int r = 0; r < 8; ++r)
          lds_vT[d][rb + r] = (_Float16)(acc[r] + bv);
      }
    }
  }
  __syncthreads();

  // ---- Stage C: attention, one head per wave ------------------------------
  {
    const int   h     = wave;
    const float scale = 0.17677669529663687f;    // 32^-0.5
    const bool  lastY = (wy == 31), lastX = (wx == 31);
    const float* rp_h = lds_rel + h * 225;

    for (int Mt = 0; Mt < 4; ++Mt) {
      // S = (q*scale) @ k^T : K = head_dim = 32 -> one WMMA per tile
      v16h aq = load_a_frag(&lds_qk[0][0], Mt * 16, QK_LD, h * HDIM, lane);
      v8f s[4];
#pragma unroll
      for (int Nt = 0; Nt < 4; ++Nt) {
        v16h bk = load_bT_frag(&lds_qk[0][0] + 256 + h * HDIM,
                               Nt * 16, QK_LD, 0, lane);
        v8f z = {};
        s[Nt] = wmma_f16(aq, bk, z);
      }
      // scale + rel-pos bias + SW mask
      const int p_base = Mt * 16 + half16;
#pragma unroll
      for (int Nt = 0; Nt < 4; ++Nt) {
        const int q  = Nt * 16 + q_n;
        const int qy = q >> 3, qx = q & 7;
#pragma unroll
        for (int r = 0; r < 8; ++r) {
          const int p  = p_base + r;
          const int py = p >> 3, px = p & 7;
          float v = s[Nt][r] * scale +
                    rp_h[(py - qy + 7) * 15 + (px - qx + 7)];
          const bool msk = (lastY && ((py < 4) != (qy < 4))) ||
                           (lastX && ((px < 4) != (qx < 4)));
          s[Nt][r] = msk ? NEG_INF : v;
        }
      }
      // softmax along q: 4 frags + 16-lane shfl_xor reduction per row
#pragma unroll
      for (int r = 0; r < 8; ++r) {
        float m = fmaxf(fmaxf(s[0][r], s[1][r]), fmaxf(s[2][r], s[3][r]));
#pragma unroll
        for (int off = 1; off < 16; off <<= 1)
          m = fmaxf(m, __shfl_xor(m, off, 32));
        float e[4], sum = 0.f;
#pragma unroll
        for (int Nt = 0; Nt < 4; ++Nt) {
          e[Nt] = __expf(s[Nt][r] - m);
          sum += e[Nt];
        }
#pragma unroll
        for (int off = 1; off < 16; off <<= 1)
          sum += __shfl_xor(sum, off, 32);
        const float inv = 1.0f / sum;
#pragma unroll
        for (int Nt = 0; Nt < 4; ++Nt) s[Nt][r] = e[Nt] * inv;
      }
      // transpose P (C-layout -> A-layout) via per-wave LDS tile
#pragma unroll
      for (int Nt = 0; Nt < 4; ++Nt) {
        const int q = Nt * 16 + q_n;
#pragma unroll
        for (int r = 0; r < 8; ++r)
          p_buf[h][half16 + r][q] = (_Float16)s[Nt][r];
      }
      // out_tile[16x32] = P[16x64] @ V[64x32]; V^T is K-contiguous in LDS
#pragma unroll
      for (int Nt = 0; Nt < 2; ++Nt) {
        v8f acc = {};
#pragma unroll
        for (int kc = 0; kc < 2; ++kc) {
          v16h ap = load_a_frag(&p_buf[h][0][0], 0, P_LD, kc * 32, lane);
          v16h bv = load_bT_frag(&lds_vT[0][0], h * HDIM + Nt * 16,
                                 VT_LD, kc * 32, lane);
          acc = wmma_f16(ap, bv, acc);
        }
        const int d  = h * HDIM + Nt * 16 + q_n;
        const int rb = Mt * 16 + half16;
#pragma unroll
        for (int r = 0; r < 8; ++r)
          lds_x[rb + r][d] = (_Float16)acc[r];   // reuse lds_x as attn-out
      }
    }
  }
  __syncthreads();

  // ---- Stage D: output projection + un-partition + roll(+4,+4) ------------
  // Reference quirk: out.transpose(1,0,2,3,4).reshape(b,...) reshuffles
  // (n_idx,b_idx) -> flat -> (b', window'). Out-proj is per-token, so we
  // just remap the destination.
  {
    const int flat = n_idx * 4 + b_idx;
    const int bp   = flat >> 10;
    const int wp   = flat & 1023;
    const int wyp = wp >> 5, wxp = wp & 31;

    const int Mt = wave >> 1;
    v16h aO[8];
#pragma unroll
    for (int kc = 0; kc < 8; ++kc)
      aO[kc] = load_a_frag(&lds_x[0][0], Mt * 16, X_LD, kc * 32, lane);

    const int tb  = Mt * 16 + half16;
    const int nt0 = (wave & 1) * 8;
    for (int Nt = nt0; Nt < nt0 + 8; ++Nt) {     // 16 N-tiles of 16
      v8f acc = {};
#pragma unroll
      for (int kc = 0; kc < 8; ++kc) {
        v16h b = load_bT_frag(wout_h, Nt * 16, 256, kc * 32, lane);
        acc = wmma_f16(aO[kc], b, acc);
      }
      const int o  = Nt * 16 + q_n;
      const float bv = b_out[o];
#pragma unroll
      for (int r = 0; r < 8; ++r) {
        const int t  = tb + r;
        const int ty = t >> 3, tx = t & 7;
        const int row = (wyp * 8 + ty + 4) & 255;
        const int col = (wxp * 8 + tx + 4) & 255;
        out[(((size_t)bp * IMG + row) * IMG + col) * CH + o] = acc[r] + bv;
      }
    }
  }
}

extern "C" void kernel_launch(void* const* d_in, const int* in_sizes, int n_in,
                              void* d_out, int out_size, void* d_ws, size_t ws_size,
                              hipStream_t stream) {
  const float* x       = (const float*)d_in[0];
  const float* w_qkv   = (const float*)d_in[1];
  const float* b_qkv   = (const float*)d_in[2];
  const float* rel_pos = (const float*)d_in[3];
  const float* w_out   = (const float*)d_in[4];
  const float* b_out   = (const float*)d_in[5];

  _Float16* wqkv_h = (_Float16*)d_ws;            // 768*256*2 = 384 KB
  _Float16* wout_h = wqkv_h + 768 * 256;         // 256*256*2 = 128 KB

  cvt_f32_f16<<<(768 * 256 + 255) / 256, 256, 0, stream>>>(w_qkv, wqkv_h, 768 * 256);
  cvt_f32_f16<<<(256 * 256 + 255) / 256, 256, 0, stream>>>(w_out, wout_h, 256 * 256);

  swin_fused<<<4096, 256, 0, stream>>>(x, wqkv_h, b_qkv, rel_pos,
                                       wout_h, b_out, (float*)d_out);
}
